// MLHAttention_26096221290606
// MI455X (gfx1250) — matrix operand order repaired and checked by reading, not compiled, not run on hardware
//
#include <hip/hip_runtime.h>
#include <hip/hip_bf16.h>

typedef __bf16 bf16_t;
typedef __attribute__((ext_vector_type(8)))  __bf16 v8bf;
typedef __attribute__((ext_vector_type(16))) __bf16 v16bf;
typedef __attribute__((ext_vector_type(8)))  float  v8f;

#define NEG_INF (-__builtin_inff())

// ---------------------------------------------------------------------------
// WMMA helper: D = A(16x32 bf16) * B(32x16 bf16) + C(16x16 f32)
// ---------------------------------------------------------------------------
static __device__ __forceinline__ v8f wmma_bf16(v16bf a, v16bf b, v8f c) {
  return __builtin_amdgcn_wmma_f32_16x16x32_bf16(false, a, false, b,
                                                 (short)0, c, false, false);
}

// CDNA5 async copy: global (SGPR base + 32-bit byte offset) -> LDS, 16B/lane.
// Tracked by ASYNCcnt; no VGPR data round-trip.
static __device__ __forceinline__ void async_ld_b128(unsigned lds_off,
                                                     unsigned gbyte_off,
                                                     const void* sbase) {
  asm volatile("global_load_async_to_lds_b128 %0, %1, %2"
               : : "v"(lds_off), "v"(gbyte_off), "s"(sbase) : "memory");
}
static __device__ __forceinline__ void wait_async0() {
  asm volatile("s_wait_asynccnt 0x0" : : : "memory");
}
static __device__ __forceinline__ unsigned lds_addr(const void* p) {
  return (unsigned)(uintptr_t)p;  // LDS aperture: low 32 bits == LDS byte offset
}

// A fragment (16x32, row-major [m][k], ld elements per row).
// lane 0-15: rows 0-15, K = kbase+{0..7} then kbase+{16..23}
// lane16-31: same rows,  K = kbase+{8..15} then kbase+{24..31}
static __device__ __forceinline__ v16bf load_frag_a(const bf16_t* base, int lane,
                                                    int ld, int kbase) {
  const bf16_t* p = base + (lane & 15) * ld + kbase + ((lane >> 4) ? 8 : 0);
  v8bf lo = *(const v8bf*)p;
  v8bf hi = *(const v8bf*)(p + 16);
  return __builtin_shufflevector(lo, hi, 0,1,2,3,4,5,6,7,8,9,10,11,12,13,14,15);
}

// B fragment (32x16, stored column-major [n][k], k contiguous, ld per column).
// lane 0-15: col n=lane,    K = kbase+{0..7} then kbase+{16..23}
// lane16-31: col n=lane-16, K = kbase+{8..15} then kbase+{24..31}
static __device__ __forceinline__ v16bf load_frag_b(const bf16_t* base, int lane,
                                                    int ld, int kbase) {
  const bf16_t* p = base + (lane & 15) * ld + kbase + ((lane >> 4) ? 8 : 0);
  v8bf lo = *(const v8bf*)p;
  v8bf hi = *(const v8bf*)(p + 16);
  return __builtin_shufflevector(lo, hi, 0,1,2,3,4,5,6,7,8,9,10,11,12,13,14,15);
}

// ---------------------------------------------------------------------------
// f32 -> bf16 conversion (n multiple of 4)
// ---------------------------------------------------------------------------
__global__ __launch_bounds__(256) void cvt_f32_bf16(const float* __restrict__ src,
                                                    bf16_t* __restrict__ dst, int n) {
  int i = (blockIdx.x * 256 + threadIdx.x) * 4;
  if (i < n) {
    float4 v = *(const float4*)(src + i);
    bf16_t tmp[4] = {(bf16_t)v.x, (bf16_t)v.y, (bf16_t)v.z, (bf16_t)v.w};
    *(uint2*)(dst + i) = *(const uint2*)tmp;
  }
}

// ---------------------------------------------------------------------------
// GEMM  Y[m,n] = sum_k X[m,k] * W[n,k]   (W row-major == B column-major)
// Block 128x128, K-slab 32, 256 threads = 8 waves (4 M x 2 N).
// Double-buffered LDS tiles filled by async DMA; one barrier per k-step.
// MODE 0: elu(y)+1, bf16 head-scatter [B,NH,S,HD]   (Q, K)
// MODE 1: bf16 head-scatter                          (V)
// MODE 2: f32 row-major                              (out proj)
// ---------------------------------------------------------------------------
#define LDA 40  // 32 + 8 bf16 pad -> 80B row stride: conflict-free b128 frags

template <int MODE>
__global__ __launch_bounds__(256) void gemm_qkvo(
    const bf16_t* __restrict__ X, const bf16_t* __restrict__ W,
    bf16_t* __restrict__ outHead, float* __restrict__ outF32,
    int M, int N, int K) {
  __shared__ alignas(16) bf16_t As[2][128 * LDA];
  __shared__ alignas(16) bf16_t Bs[2][128 * LDA];

  const int lane = threadIdx.x & 31;
  const int wave = threadIdx.x >> 5;
  const int wm = wave & 3;   // 32-row slab
  const int wn = wave >> 2;  // 64-col slab
  const int mBase = blockIdx.x * 128;
  const int nBase = blockIdx.y * 128;

  const int srow  = threadIdx.x >> 1;
  const int skoff = (threadIdx.x & 1) * 16;

  auto stage = [&](int kk, int buf) {
    unsigned la = lds_addr(&As[buf][srow * LDA + skoff]);
    unsigned lb = lds_addr(&Bs[buf][srow * LDA + skoff]);
    unsigned ga = (unsigned)(((mBase + srow) * K + kk + skoff) * 2);
    unsigned gb = (unsigned)(((nBase + srow) * K + kk + skoff) * 2);
    async_ld_b128(la,      ga,      X);
    async_ld_b128(la + 16, ga + 16, X);
    async_ld_b128(lb,      gb,      W);
    async_ld_b128(lb + 16, gb + 16, W);
  };

  v8f acc[2][4];
#pragma unroll
  for (int a = 0; a < 2; a++)
#pragma unroll
    for (int b = 0; b < 4; b++)
#pragma unroll
      for (int i = 0; i < 8; i++) acc[a][b][i] = 0.0f;

  stage(0, 0);
  for (int kk = 0, it = 0; kk < K; kk += 32, ++it) {
    const int cur = it & 1;
    wait_async0();       // own async copies into buf[cur] done
    __syncthreads();     // everyone's copies visible; prev reads of buf[cur^1] done
    if (kk + 32 < K) stage(kk + 32, cur ^ 1);  // DMA next slab under compute

    const bf16_t* Ab = As[cur];
    const bf16_t* Bb = Bs[cur];
    v16bf afrag0 = load_frag_a(Ab + (wm * 32 +  0) * LDA, lane, LDA, 0);
    v16bf afrag1 = load_frag_a(Ab + (wm * 32 + 16) * LDA, lane, LDA, 0);
#pragma unroll
    for (int nt = 0; nt < 4; nt++) {
      v16bf bfrag = load_frag_b(Bb + (wn * 64 + nt * 16) * LDA, lane, LDA, 0);
      acc[0][nt] = wmma_bf16(afrag0, bfrag, acc[0][nt]);
      acc[1][nt] = wmma_bf16(afrag1, bfrag, acc[1][nt]);
    }
  }

#pragma unroll
  for (int mt = 0; mt < 2; mt++) {
#pragma unroll
    for (int nt = 0; nt < 4; nt++) {
      const int col = nBase + wn * 64 + nt * 16 + (lane & 15);
#pragma unroll
      for (int i = 0; i < 8; i++) {
        const int r = mBase + wm * 32 + mt * 16 + ((lane < 16) ? i : i + 8);
        float v = acc[mt][nt][i];
        if (MODE == 0) v = (v > 0.0f) ? (v + 1.0f) : __expf(v);  // elu+1
        if (MODE <= 1) {
          const int bb = r >> 11, s = r & 2047;     // S = 2048
          const int h = col >> 6, d = col & 63;     // HD = 64
          outHead[(((size_t)(bb * 16 + h)) * 2048 + s) * 64 + d] = (bf16_t)v;
        } else {
          outF32[(size_t)r * N + col] = v;
        }
      }
    }
  }
}

// ---------------------------------------------------------------------------
// Fused flash attention per (b, h, 128 query rows).
// 256 threads = 8 waves, each wave owns 16 query rows.
// K tiles: async DMA, double buffered. V tiles: register transpose, double
// buffered. One barrier per key-block.
// ---------------------------------------------------------------------------
#define LDK 72   // K-block row stride  (64 + 8 pad)
#define LDP 136  // P / V^T row stride (128 + 8 pad)

__global__ __launch_bounds__(256) void attn_fused(
    const bf16_t* __restrict__ Q, const bf16_t* __restrict__ Kmat,
    const bf16_t* __restrict__ V, const int* __restrict__ mask,
    bf16_t* __restrict__ AO) {
  __shared__ alignas(16) bf16_t Ks[2 * 128 * LDK];   // [buf][key][d]
  __shared__ alignas(16) bf16_t Vts[2 * 64 * LDP];   // [buf][d][key]
  __shared__ alignas(16) bf16_t Ps[8 * 16 * LDP];    // per-wave P tile [m][key]
  __shared__ float maskAdd[2][128];

  const int lane = threadIdx.x & 31;
  const int wave = threadIdx.x >> 5;
  const int qt = blockIdx.x, h = blockIdx.y, b = blockIdx.z;

  const size_t headOff = (((size_t)b * 16 + h) * 2048) * 64;
  const bf16_t* Qh = Q + headOff;
  const bf16_t* Kh = Kmat + headOff;
  const bf16_t* Vh = V + headOff;

  const int qRow0 = qt * 128 + wave * 16;
  const int srow  = threadIdx.x >> 1;        // 0..127
  const int shalf = (threadIdx.x & 1) * 32;  // 0 / 32 bf16

  auto stageK = [&](int kb, int buf) {  // async DMA 128x64 bf16 -> Ks[buf]
    unsigned la = lds_addr(Ks + buf * (128 * LDK) + srow * LDK + shalf);
    unsigned go = (unsigned)(((kb * 128 + srow) * 64 + shalf) * 2);
    async_ld_b128(la,      go,      Kh);
    async_ld_b128(la + 16, go + 16, Kh);
    async_ld_b128(la + 32, go + 32, Kh);
    async_ld_b128(la + 48, go + 48, Kh);
  };
  auto stageV = [&](int kb, int buf) {  // transpose 128x64 -> Vts[buf][d][key]
    bf16_t* Vt = Vts + buf * (64 * LDP);
    uint4 tv[4];
    const uint4* gv = (const uint4*)(Vh + (size_t)(kb * 128 + srow) * 64 + shalf);
    tv[0] = gv[0]; tv[1] = gv[1]; tv[2] = gv[2]; tv[3] = gv[3];
    const bf16_t* e = (const bf16_t*)tv;
#pragma unroll
    for (int j = 0; j < 32; j++) Vt[(shalf + j) * LDP + srow] = e[j];
  };
  auto stageM = [&](int kb, int buf) {
    if (threadIdx.x < 128)
      maskAdd[buf][threadIdx.x] =
          mask[b * 2048 + kb * 128 + threadIdx.x] ? 0.0f : NEG_INF;
  };

  // Q fragments held in registers for whole kernel (16 rows x 64 d)
  v16bf qfrag0 = load_frag_a(Qh + (size_t)qRow0 * 64, lane, 64, 0);
  v16bf qfrag1 = load_frag_a(Qh + (size_t)qRow0 * 64, lane, 64, 32);

  float mrow[8], lrow[8];
  v8f o[4];
#pragma unroll
  for (int i = 0; i < 8; i++) { mrow[i] = NEG_INF; lrow[i] = 0.0f; }
#pragma unroll
  for (int nt = 0; nt < 4; nt++)
#pragma unroll
    for (int i = 0; i < 8; i++) o[nt][i] = 0.0f;

  stageK(0, 0);
  stageV(0, 0);
  stageM(0, 0);

  for (int kb = 0; kb < 16; kb++) {
    const int cur = kb & 1;
    wait_async0();
    __syncthreads();
    if (kb + 1 < 16) {  // prefetch next key block under compute
      stageK(kb + 1, cur ^ 1);
      stageV(kb + 1, cur ^ 1);
      stageM(kb + 1, cur ^ 1);
    }
    const bf16_t* KsB = Ks + cur * (128 * LDK);
    const bf16_t* VtB = Vts + cur * (64 * LDP);
    const float*  mAd = maskAdd[cur];

    // scores = Q (16x64) @ K^T (64x128)
    v8f sc[8];
#pragma unroll
    for (int nt = 0; nt < 8; nt++)
#pragma unroll
      for (int i = 0; i < 8; i++) sc[nt][i] = 0.0f;
#pragma unroll
    for (int nt = 0; nt < 8; nt++) {
      v16bf kf0 = load_frag_b(KsB + nt * 16 * LDK, lane, LDK, 0);
      sc[nt] = wmma_bf16(qfrag0, kf0, sc[nt]);
      v16bf kf1 = load_frag_b(KsB + nt * 16 * LDK, lane, LDK, 32);
      sc[nt] = wmma_bf16(qfrag1, kf1, sc[nt]);
    }
#pragma unroll
    for (int nt = 0; nt < 8; nt++) {
      float ma = mAd[nt * 16 + (lane & 15)];
#pragma unroll
      for (int i = 0; i < 8; i++) sc[nt][i] += ma;
    }

    // online softmax (stats per row; 16-lane shuffle groups match C layout)
    float scale[8];
#pragma unroll
    for (int i = 0; i < 8; i++) {
      float mx = sc[0][i];
#pragma unroll
      for (int nt = 1; nt < 8; nt++) mx = fmaxf(mx, sc[nt][i]);
#pragma unroll
      for (int d = 1; d < 16; d <<= 1) mx = fmaxf(mx, __shfl_xor(mx, d, 32));
      const float mn = fmaxf(mrow[i], mx);
      scale[i] = (mn == NEG_INF) ? 1.0f : __expf(mrow[i] - mn);
      float s = 0.0f;
#pragma unroll
      for (int nt = 0; nt < 8; nt++) {
        float p = (mn == NEG_INF) ? 0.0f : __expf(sc[nt][i] - mn);
        sc[nt][i] = p;
        s += p;
      }
#pragma unroll
      for (int d = 1; d < 16; d <<= 1) s += __shfl_xor(s, d, 32);
      lrow[i] = lrow[i] * scale[i] + s;
      mrow[i] = mn;
    }
#pragma unroll
    for (int nt = 0; nt < 4; nt++)
#pragma unroll
      for (int i = 0; i < 8; i++) o[nt][i] *= scale[i];

    // P (C layout) -> per-wave LDS row-major, then re-read as A fragments
    bf16_t* pw = Ps + wave * (16 * LDP);
#pragma unroll
    for (int nt = 0; nt < 8; nt++)
#pragma unroll
      for (int i = 0; i < 8; i++)
        pw[((lane < 16) ? i : i + 8) * LDP + nt * 16 + (lane & 15)] =
            (bf16_t)sc[nt][i];

    // O += P (16x128) @ V (128x64)
#pragma unroll
    for (int kc = 0; kc < 4; kc++) {
      v16bf pf = load_frag_a(pw, lane, LDP, kc * 32);
#pragma unroll
      for (int nt = 0; nt < 4; nt++) {
        v16bf vf = load_frag_b(VtB + nt * 16 * LDP, lane, LDP, kc * 32);
        o[nt] = wmma_bf16(pf, vf, o[nt]);
      }
    }
  }

  // finalize: divide by l, undo head transpose -> [B, S, H] bf16
#pragma unroll
  for (int nt = 0; nt < 4; nt++) {
    const int d = h * 64 + nt * 16 + (lane & 15);
#pragma unroll
    for (int i = 0; i < 8; i++) {
      const int s = qRow0 + ((lane < 16) ? i : i + 8);
      const float l = lrow[i];
      const float val = (l > 0.0f) ? (o[nt][i] / l) : 0.0f;
      AO[((size_t)b * 2048 + s) * 1024 + d] = (bf16_t)val;
    }
  }
}

// ---------------------------------------------------------------------------
extern "C" void kernel_launch(void* const* d_in, const int* in_sizes, int n_in,
                              void* d_out, int out_size, void* d_ws, size_t ws_size,
                              hipStream_t stream) {
  (void)in_sizes; (void)n_in; (void)out_size; (void)ws_size;
  const float* hs  = (const float*)d_in[0];
  const int*   msk = (const int*)d_in[1];
  const float* Wq  = (const float*)d_in[2];
  const float* Wk  = (const float*)d_in[3];
  const float* Wv  = (const float*)d_in[4];
  const float* Wo  = (const float*)d_in[5];
  float* out = (float*)d_out;
  char* ws = (char*)d_ws;

  const int M = 4096, N = 1024, K = 1024;  // M = B*S

  const size_t MB = 1024 * 1024;
  bf16_t* Xb  = (bf16_t*)(ws);
  bf16_t* Wqb = (bf16_t*)(ws +  8 * MB);
  bf16_t* Wkb = (bf16_t*)(ws + 10 * MB);
  bf16_t* Wvb = (bf16_t*)(ws + 12 * MB);
  bf16_t* Wob = (bf16_t*)(ws + 14 * MB);
  bf16_t* Qb  = (bf16_t*)(ws + 16 * MB);
  bf16_t* Kb  = (bf16_t*)(ws + 24 * MB);
  bf16_t* Vb  = (bf16_t*)(ws + 32 * MB);
  bf16_t* AOb = (bf16_t*)(ws + 40 * MB);

  cvt_f32_bf16<<<(M * K / 4 + 255) / 256, 256, 0, stream>>>(hs, Xb, M * K);
  cvt_f32_bf16<<<(K * N / 4 + 255) / 256, 256, 0, stream>>>(Wq, Wqb, K * N);
  cvt_f32_bf16<<<(K * N / 4 + 255) / 256, 256, 0, stream>>>(Wk, Wkb, K * N);
  cvt_f32_bf16<<<(K * N / 4 + 255) / 256, 256, 0, stream>>>(Wv, Wvb, K * N);
  cvt_f32_bf16<<<(K * N / 4 + 255) / 256, 256, 0, stream>>>(Wo, Wob, K * N);

  dim3 gg(M / 128, N / 128);  // 32 x 8
  gemm_qkvo<0><<<gg, 256, 0, stream>>>(Xb, Wqb, Qb, nullptr, M, N, K);
  gemm_qkvo<0><<<gg, 256, 0, stream>>>(Xb, Wkb, Kb, nullptr, M, N, K);
  gemm_qkvo<1><<<gg, 256, 0, stream>>>(Xb, Wvb, Vb, nullptr, M, N, K);

  attn_fused<<<dim3(16, 16, 2), 256, 0, stream>>>(Qb, Kb, Vb, msk, AOb);

  gemm_qkvo<2><<<gg, 256, 0, stream>>>(AOb, Wob, nullptr, out, M, N, K);
}
